// LiquidRecurrent_24807731101747
// MI455X (gfx1250) — matrix operand order repaired and checked by reading, not compiled, not run on hardware
//
#include <hip/hip_runtime.h>
#include <math.h>

typedef __attribute__((ext_vector_type(16))) _Float16 v16h;
typedef __attribute__((ext_vector_type(8)))  float    v8f;

#define T_DIM   512
#define C_DIM   128
#define N_DIM   256
#define O_DIM   64
#define B_DIM   256
#define UNFOLD  6
#define BT      16      // batch rows per workgroup
#define NWAVES  16      // one 16-wide N-slice per wave
#define XSTR    264     // padded LDS row stride (halfs): 528B = 33*16B -> conflict-free b128

// Fast transcendental helpers: single hardware transcendental ops instead of
// ocml polynomial tanh and IEEE div (these sit on the sequential recurrence
// critical path between dependent WMMAs).
__device__ __forceinline__ float fast_rcp(float x) {
  return __builtin_amdgcn_rcpf(x);
}
__device__ __forceinline__ float fast_tanh(float x) {
  // tanh(x) = 1 - 2 / (e^{2x} + 1);  e^{2x} = exp2(x * 2*log2(e))
  const float e2x = __builtin_amdgcn_exp2f(x * 2.8853900817779268f);
  return 1.0f - 2.0f * __builtin_amdgcn_rcpf(e2x + 1.0f);
}

// ---------------------------------------------------------------------------
// A-operand fragment (16x32 f16): lane = row M (lmod); this lane's K values are
// two contiguous 8-half runs starting at kbase and kbase+16 (kbase already
// includes lhalf*8). Contiguous 16B-aligned -> compiler merges to b128 loads.
// ---------------------------------------------------------------------------
template <typename PTR>
__device__ __forceinline__ v16h load_a_frag(PTR p0) {
  v16h r;
#pragma unroll
  for (int e = 0; e < 8; ++e) r[e] = p0[e];
#pragma unroll
  for (int e = 0; e < 8; ++e) r[8 + e] = p0[16 + e];
  return r;
}

// ---------------------------------------------------------------------------
// B-operand fragment (32x16 f16) from a row-major f32 weight W[n][k]:
// element (K=k, N=n) = W[n][k]. This lane holds column n, K values are two
// 8-wide runs at kbase and kbase+16 where kbase includes lhalf*8; pairs packed along VGPRs.
// ---------------------------------------------------------------------------
__device__ __forceinline__ v16h load_b_frag(const float* __restrict__ W,
                                            int n, int rowlen, int kbase) {
  v16h r;
  const float* p = W + (size_t)n * rowlen + kbase;
#pragma unroll
  for (int e = 0; e < 8; ++e) r[e] = (_Float16)p[e];
#pragma unroll
  for (int e = 0; e < 8; ++e) r[8 + e] = (_Float16)p[16 + e];
  return r;
}

// ---------------------------------------------------------------------------
// Transpose+downconvert: i (B,C,T) f32 -> ws (B,T,C) f16, tiled 32x32.
// ---------------------------------------------------------------------------
__global__ void ltc_transpose_kernel(const float* __restrict__ i_in,
                                     _Float16* __restrict__ o) {
  __shared__ _Float16 tile[32][33];
  const int b  = blockIdx.z;
  const int t0 = blockIdx.x * 32;
  const int c0 = blockIdx.y * 32;
  const int tx = threadIdx.x;          // along T on read, along C on write
#pragma unroll
  for (int cy = threadIdx.y; cy < 32; cy += 8)
    tile[cy][tx] = (_Float16)i_in[((size_t)b * C_DIM + (c0 + cy)) * T_DIM + t0 + tx];
  __syncthreads();
#pragma unroll
  for (int ty = threadIdx.y; ty < 32; ty += 8)
    o[((size_t)b * T_DIM + (t0 + ty)) * C_DIM + c0 + tx] = tile[tx][ty];
}

// ---------------------------------------------------------------------------
// Persistent recurrent kernel: 16 workgroups x 512 threads (16 wave32).
// Workgroup -> 16 batch rows; wave w -> hidden columns [16w, 16w+16).
// ---------------------------------------------------------------------------
__global__ __launch_bounds__(512, 1)
void ltc_recurrent_kernel(const _Float16* __restrict__ inp_tc,  // (B,T,C) f16
                          const float* __restrict__ delta_t,    // (B,T)
                          const float* __restrict__ x0,         // (B,N)
                          const float* __restrict__ w_rec,      // (N,N)
                          const float* __restrict__ w_in,       // (N,C)
                          const float* __restrict__ bias,       // (N)
                          const float* __restrict__ a_vec,      // (N)
                          const float* __restrict__ tau,        // (N)
                          const float* __restrict__ w_out,      // (O,N)
                          const float* __restrict__ b_out,      // (O)
                          float* __restrict__ out)              // (B,T,O)
{
  __shared__ _Float16 x_h[2][BT * XSTR];   // double-buffered f16 state
  __shared__ float dt_s[BT];

  const int tid   = threadIdx.x;
  const int wid   = tid >> 5;
  const int lane  = tid & 31;
  const int lmod  = lane & 15;
  const int lhalf = lane >> 4;
  const int b0    = blockIdx.x * BT;
  const int n     = wid * 16 + lmod;       // owned hidden column

  // per-lane elementwise constants (indexed by n)
  const float a_n  = a_vec[n];
  const float it_n = fast_rcp(tau[n]);
  const float b_n  = bias[n];

  // ---- loop-invariant weight fragments, resident in VGPRs ----
  v16h wrec_f[8];
#pragma unroll
  for (int kt = 0; kt < 8; ++kt)
    wrec_f[kt] = load_b_frag(w_rec, n, N_DIM, kt * 32 + lhalf * 8);

  v16h win_f[4];
#pragma unroll
  for (int ct = 0; ct < 4; ++ct)
    win_f[ct] = load_b_frag(w_in, n, C_DIM, ct * 32 + lhalf * 8);

  v16h wout_f[8] = {};
  float bo_n = 0.0f;
  if (wid < 4) {                            // waves 0..3 own the O=64 head
    const int o = wid * 16 + lmod;
#pragma unroll
    for (int kt = 0; kt < 8; ++kt)
      wout_f[kt] = load_b_frag(w_out, o, N_DIM, kt * 32 + lhalf * 8);
    bo_n = b_out[o];
  }

  // ---- initial state ----
  float xr[8];
#pragma unroll
  for (int r = 0; r < 8; ++r) {
    const int m = r + lhalf * 8;
    xr[r] = x0[(size_t)(b0 + m) * N_DIM + n];
  }
  int cur = 0;
#pragma unroll
  for (int r = 0; r < 8; ++r) {
    const int m = r + lhalf * 8;
    x_h[cur][m * XSTR + n] = (_Float16)xr[r];
  }
  __syncthreads();

  const _Float16* inp_row = inp_tc + (size_t)(b0 + lmod) * T_DIM * C_DIM;

  for (int t = 0; t < T_DIM; ++t) {
    if (tid < BT) dt_s[tid] = delta_t[(size_t)(b0 + tid) * T_DIM + t];

    // ---- pre_in = inp_t @ w_in^T + b  (4 WMMAs) ----
    // Load all four A-fragments first so the 8 global_load_b128 issue as one
    // clause and drain with partial loadcnt waits instead of serializing
    // load->wmma->load->wmma.
    const _Float16* irow = inp_row + (size_t)t * C_DIM;
    v16h ain[4];
#pragma unroll
    for (int ct = 0; ct < 4; ++ct)
      ain[ct] = load_a_frag(irow + ct * 32 + lhalf * 8);
    if (t + 1 < T_DIM) __builtin_prefetch(irow + C_DIM, 0, 1);  // next timestep row

    v8f pre;
#pragma unroll
    for (int r = 0; r < 8; ++r) pre[r] = b_n;
#pragma unroll
    for (int ct = 0; ct < 4; ++ct)
      pre = __builtin_amdgcn_wmma_f32_16x16x32_f16(false, ain[ct], false, win_f[ct],
                                                   (short)0, pre, false, false);

    __syncthreads();                         // dt_s visible
    float dtv[8];
#pragma unroll
    for (int r = 0; r < 8; ++r) dtv[r] = dt_s[r + lhalf * 8];

    // ---- fused ODE solver unfolding (6 x 8 WMMAs) ----
    for (int u = 0; u < UNFOLD; ++u) {
      v8f z = pre;
      const _Float16* xrow = &x_h[cur][lmod * XSTR];
#pragma unroll
      for (int kt = 0; kt < 8; ++kt) {
        v16h a = load_a_frag(xrow + kt * 32 + lhalf * 8);
        z = __builtin_amdgcn_wmma_f32_16x16x32_f16(false, a, false, wrec_f[kt],
                                                   (short)0, z, false, false);
      }
#pragma unroll
      for (int r = 0; r < 8; ++r) {
        const float f   = fast_tanh(z[r]);
        const float num = fmaf(dtv[r] * f, a_n, xr[r]);
        const float den = fmaf(dtv[r], it_n + f, 1.0f);
        xr[r] = num * fast_rcp(den);
      }
      // write next buffer (previous readers of it are >=1 barrier behind)
#pragma unroll
      for (int r = 0; r < 8; ++r) {
        const int m = r + lhalf * 8;
        x_h[cur ^ 1][m * XSTR + n] = (_Float16)xr[r];
      }
      __syncthreads();
      cur ^= 1;
    }

    // ---- output head: out_t = x @ w_out^T + b_out (waves 0..3) ----
    if (wid < 4) {
      v8f oa;
#pragma unroll
      for (int r = 0; r < 8; ++r) oa[r] = bo_n;
      const _Float16* xrow = &x_h[cur][lmod * XSTR];
#pragma unroll
      for (int kt = 0; kt < 8; ++kt) {
        v16h a = load_a_frag(xrow + kt * 32 + lhalf * 8);
        oa = __builtin_amdgcn_wmma_f32_16x16x32_f16(false, a, false, wout_f[kt],
                                                    (short)0, oa, false, false);
      }
      const int o = wid * 16 + lmod;
#pragma unroll
      for (int r = 0; r < 8; ++r) {
        const int m = r + lhalf * 8;
        out[((size_t)(b0 + m) * T_DIM + t) * O_DIM + o] = oa[r];
      }
    }
  }
}

// ---------------------------------------------------------------------------
extern "C" void kernel_launch(void* const* d_in, const int* in_sizes, int n_in,
                              void* d_out, int out_size, void* d_ws, size_t ws_size,
                              hipStream_t stream) {
  const float* i_in    = (const float*)d_in[0];  // (B,C,T)
  const float* delta_t = (const float*)d_in[1];  // (B,T)
  const float* x0      = (const float*)d_in[2];  // (B,N)
  const float* w_rec   = (const float*)d_in[3];  // (N,N)
  const float* w_in    = (const float*)d_in[4];  // (N,C)
  const float* bias    = (const float*)d_in[5];  // (N)
  const float* a_vec   = (const float*)d_in[6];  // (N)
  const float* tau     = (const float*)d_in[7];  // (N)
  const float* w_out   = (const float*)d_in[8];  // (O,N)
  const float* b_out   = (const float*)d_in[9];  // (O)
  float* out = (float*)d_out;                    // (B,T,O)

  _Float16* inp_tc = (_Float16*)d_ws;            // (B,T,C) f16, 32 MB

  dim3 tg(T_DIM / 32, C_DIM / 32, B_DIM);
  ltc_transpose_kernel<<<tg, dim3(32, 8), 0, stream>>>(i_in, inp_tc);

  ltc_recurrent_kernel<<<B_DIM / BT, NWAVES * 32, 0, stream>>>(
      inp_tc, delta_t, x0, w_rec, w_in, bias, a_vec, tau, w_out, b_out, out);
}
